// SimpleGATv2_25692494365527
// MI455X (gfx1250) — matrix operand order repaired and checked by reading, not compiled, not run on hardware
//
#include <hip/hip_runtime.h>
#include <hip/hip_bf16.h>

// ---------------- problem constants ----------------
#define NN      100000      // nodes
#define EE      1600000     // raw edges
#define E2      1700000     // edges + self loops
#define F_IN    128
#define HID     32
#define HEADS   4
#define CLASSES 10
#define NEG_SLOPE 0.2f

typedef __attribute__((ext_vector_type(16))) __bf16 v16bf;
typedef __attribute__((ext_vector_type(8)))  float  v8f;

// ---------------- helpers ----------------
__device__ __forceinline__ float lrelu(float v) { return v > 0.f ? v : NEG_SLOPE * v; }

// monotonic uint key for float atomicMax (total order incl. negatives)
__device__ __forceinline__ unsigned fkey(float f) {
    unsigned u = __float_as_uint(f);
    return (u & 0x80000000u) ? ~u : (u | 0x80000000u);
}
__device__ __forceinline__ float keyf(unsigned k) {
    unsigned u = (k & 0x80000000u) ? (k & 0x7fffffffu) : ~k;
    return __uint_as_float(u);
}

// =====================================================================
// Pack W[K, ncols] (fp32 row-major) into WMMA-B fragment order, bf16,
// zero-padded to 16-column tiles.  Element i (0..15) of lane L's fragment
// for (ntile, kstep) is  W[kstep*32 + (L>>4)*16 + i][ntile*16 + (L&15)].
// Flat layout: Bp[((nt*ksteps + ks)*32 + lane)*16 + i]  -> each lane's
// fragment is 32 contiguous bytes (two b128 loads, no divergence).
// =====================================================================
__global__ __launch_bounds__(256)
void pack_B(const float* __restrict__ W, __bf16* __restrict__ Bp,
            int ncols, int ntiles, int ksteps) {
    const int idx = blockIdx.x * blockDim.x + threadIdx.x;
    const int total = ntiles * ksteps * 32 * 16;
    if (idx >= total) return;
    const int i    = idx & 15;
    const int lane = (idx >> 4) & 31;
    const int ks   = (idx >> 9) % ksteps;
    const int nt   = (idx >> 9) / ksteps;
    const int k = ks * 32 + (lane >> 4) * 16 + i;
    const int n = nt * 16 + (lane & 15);
    const float v = (n < ncols) ? W[k * ncols + n] : 0.f;
    Bp[idx] = (__bf16)v;
}

// =====================================================================
// GEMM: C[M,ldc] = A[M,K] x B (packed bf16 fragments).  One wave per
// 16-row M-tile computing NTILES 16x16 tiles.  Fully unrolled over K so
// all B loads use immediate offsets; per K-step, all NTILES B fragments
// are preloaded (one clause, one wait) before the WMMA burst, and no
// operand register is rewritten while a WMMA may still read it.
// =====================================================================
template<int NTILES, int K>
__global__ __launch_bounds__(256)
void gemm_bf16_wmma(const float* __restrict__ A, const v16bf* __restrict__ Bp,
                    float* __restrict__ C, int M, int ncols, int ldc) {
    constexpr int KS = K / 32;
    const int wave  = blockIdx.x * (blockDim.x >> 5) + (threadIdx.x >> 5);
    const int lane  = threadIdx.x & 31;
    const int mtile = wave;
    if (mtile * 16 >= M) return;                 // wave-uniform guard

    const int half = lane >> 4;                  // 0 or 1
    const int mrow = mtile * 16 + (lane & 15);   // A row for this lane
    const float* __restrict__ arow = A + mrow * K + half * 16;
    const v16bf* __restrict__ bp   = Bp + lane;  // fragment index strides by 32

    v8f acc[NTILES];
    #pragma unroll
    for (int nt = 0; nt < NTILES; ++nt) acc[nt] = (v8f){};

    #pragma unroll
    for (int ks = 0; ks < KS; ++ks) {
        // ---- preload all B fragments for this K-step (immediate offsets) ----
        v16bf bfrag[NTILES];
        #pragma unroll
        for (int nt = 0; nt < NTILES; ++nt)
            bfrag[nt] = bp[(nt * KS + ks) * 32];

        // ---- A fragment: one contiguous 16-float segment -> 4x b128 ----
        const float4 f0 = *reinterpret_cast<const float4*>(arow + ks * 32);
        const float4 f1 = *reinterpret_cast<const float4*>(arow + ks * 32 + 4);
        const float4 f2 = *reinterpret_cast<const float4*>(arow + ks * 32 + 8);
        const float4 f3 = *reinterpret_cast<const float4*>(arow + ks * 32 + 12);
        v16bf a;
        a[0]  = (__bf16)f0.x; a[1]  = (__bf16)f0.y; a[2]  = (__bf16)f0.z; a[3]  = (__bf16)f0.w;
        a[4]  = (__bf16)f1.x; a[5]  = (__bf16)f1.y; a[6]  = (__bf16)f1.z; a[7]  = (__bf16)f1.w;
        a[8]  = (__bf16)f2.x; a[9]  = (__bf16)f2.y; a[10] = (__bf16)f2.z; a[11] = (__bf16)f2.w;
        a[12] = (__bf16)f3.x; a[13] = (__bf16)f3.y; a[14] = (__bf16)f3.z; a[15] = (__bf16)f3.w;

        // ---- WMMA burst: no operand rewrites in between ----
        #pragma unroll
        for (int nt = 0; nt < NTILES; ++nt)
            acc[nt] = __builtin_amdgcn_wmma_f32_16x16x32_bf16(
                false, a, false, bfrag[nt], (short)0, acc[nt], false, false);
    }

    #pragma unroll
    for (int nt = 0; nt < NTILES; ++nt) {
        const int ncol = nt * 16 + (lane & 15);
        if (ncol < ncols) {
            #pragma unroll
            for (int r = 0; r < 8; ++r) {
                const int mm = mtile * 16 + r + 8 * half;  // C/D: VGPR r -> M=r+8*half
                C[mm * ldc + ncol] = acc[nt][r];
            }
        }
    }
}

// =====================================================================
// Layer-1 edge scoring: one wave32 per edge; 4 heads x 32 ch = 4 floats/lane.
// score[e][h] = att1[h] . lrelu(xl[src] + xr[dst]); atomicMax per (dst,h).
// =====================================================================
__global__ __launch_bounds__(256)
void edge_score_l1(const int* __restrict__ ei, const float* __restrict__ xl,
                   const float* __restrict__ xr, const float* __restrict__ att,
                   float* __restrict__ score, unsigned* __restrict__ mkey) {
    const int wave = blockIdx.x * (blockDim.x >> 5) + (threadIdx.x >> 5);
    if (wave >= E2) return;
    const int lane = threadIdx.x & 31;
    int s, d;
    if (wave < EE) { s = ei[wave]; d = ei[EE + wave]; }
    else           { s = d = wave - EE; }

    const int h  = lane >> 3;
    const int c0 = (lane & 7) * 4;
    const float4 a = *reinterpret_cast<const float4*>(&xl[(size_t)s * 128 + h * 32 + c0]);
    const float4 b = *reinterpret_cast<const float4*>(&xr[(size_t)d * 128 + h * 32 + c0]);
    const float* at = &att[h * 32 + c0];

    float p = lrelu(a.x + b.x) * at[0] + lrelu(a.y + b.y) * at[1] +
              lrelu(a.z + b.z) * at[2] + lrelu(a.w + b.w) * at[3];
    p += __shfl_xor(p, 1, 32);
    p += __shfl_xor(p, 2, 32);
    p += __shfl_xor(p, 4, 32);               // sum over 8 lanes of this head
    if ((lane & 7) == 0) {
        score[(size_t)wave * HEADS + h] = p;
        atomicMax(&mkey[(size_t)d * HEADS + h], fkey(p));
    }
}

// =====================================================================
// Layer-1 aggregation (unnormalized): w = exp(score - max); denom += w;
// hacc[dst] += w * xl[src].  Normalization happens in node pass.
// =====================================================================
__global__ __launch_bounds__(256)
void edge_agg_l1(const int* __restrict__ ei, const float* __restrict__ xl,
                 const float* __restrict__ score, const unsigned* __restrict__ mkey,
                 float* __restrict__ denom, float* __restrict__ hacc) {
    const int wave = blockIdx.x * (blockDim.x >> 5) + (threadIdx.x >> 5);
    if (wave >= E2) return;
    const int lane = threadIdx.x & 31;
    int s, d;
    if (wave < EE) { s = ei[wave]; d = ei[EE + wave]; }
    else           { s = d = wave - EE; }

    const int h  = lane >> 3;
    const int c0 = (lane & 7) * 4;
    const float m = keyf(mkey[(size_t)d * HEADS + h]);
    const float w = __expf(score[(size_t)wave * HEADS + h] - m);
    if ((lane & 7) == 0) atomicAdd(&denom[(size_t)d * HEADS + h], w);

    const float4 a = *reinterpret_cast<const float4*>(&xl[(size_t)s * 128 + h * 32 + c0]);
    float* o = &hacc[(size_t)d * 128 + h * 32 + c0];
    atomicAdd(o + 0, w * a.x);
    atomicAdd(o + 1, w * a.y);
    atomicAdd(o + 2, w * a.z);
    atomicAdd(o + 3, w * a.w);
}

// normalize + bias + relu, in place: h = relu(hacc/denom + b1)
__global__ __launch_bounds__(256)
void node_norm_l1(float* __restrict__ h, const float* __restrict__ denom,
                  const float* __restrict__ b1) {
    const size_t idx = (size_t)blockIdx.x * blockDim.x + threadIdx.x;
    if (idx >= (size_t)NN * 128) return;
    const int n   = (int)(idx >> 7);
    const int col = (int)(idx & 127);
    const int hd  = col >> 5;
    float v = h[idx] / denom[(size_t)n * HEADS + hd] + b1[col];
    h[idx] = v > 0.f ? v : 0.f;
}

// =====================================================================
// Layer-2 (1 head, 10 channels): thread-per-edge variants.
// =====================================================================
__global__ __launch_bounds__(256)
void edge_score_l2(const int* __restrict__ ei, const float* __restrict__ xl,
                   const float* __restrict__ xr, const float* __restrict__ att,
                   float* __restrict__ score, unsigned* __restrict__ mkey) {
    const int e = blockIdx.x * blockDim.x + threadIdx.x;
    if (e >= E2) return;
    int s, d;
    if (e < EE) { s = ei[e]; d = ei[EE + e]; }
    else        { s = d = e - EE; }
    float p = 0.f;
    #pragma unroll
    for (int c = 0; c < CLASSES; ++c)
        p += lrelu(xl[(size_t)s * CLASSES + c] + xr[(size_t)d * CLASSES + c]) * att[c];
    score[e] = p;
    atomicMax(&mkey[d], fkey(p));
}

__global__ __launch_bounds__(256)
void edge_agg_l2(const int* __restrict__ ei, const float* __restrict__ xl,
                 const float* __restrict__ score, const unsigned* __restrict__ mkey,
                 float* __restrict__ denom, float* __restrict__ out) {
    const int e = blockIdx.x * blockDim.x + threadIdx.x;
    if (e >= E2) return;
    int s, d;
    if (e < EE) { s = ei[e]; d = ei[EE + e]; }
    else        { s = d = e - EE; }
    const float w = __expf(score[e] - keyf(mkey[d]));
    atomicAdd(&denom[d], w);
    #pragma unroll
    for (int c = 0; c < CLASSES; ++c)
        atomicAdd(&out[(size_t)d * CLASSES + c], w * xl[(size_t)s * CLASSES + c]);
}

__global__ __launch_bounds__(256)
void node_final(float* __restrict__ out, const float* __restrict__ denom,
                const float* __restrict__ b2) {
    const size_t idx = (size_t)blockIdx.x * blockDim.x + threadIdx.x;
    if (idx >= (size_t)NN * CLASSES) return;
    const int n = (int)(idx / CLASSES);
    const int c = (int)(idx % CLASSES);
    out[idx] = out[idx] / denom[n] + b2[c];
}

// =====================================================================
// launch
// =====================================================================
static inline size_t align256(size_t x) { return (x + 255) & ~(size_t)255; }

extern "C" void kernel_launch(void* const* d_in, const int* in_sizes, int n_in,
                              void* d_out, int out_size, void* d_ws, size_t ws_size,
                              hipStream_t stream) {
    const float* x    = (const float*)d_in[0];
    const int*   ei   = (const int*)  d_in[1];
    const float* W1l  = (const float*)d_in[2];
    const float* W1r  = (const float*)d_in[3];
    const float* att1 = (const float*)d_in[4];
    const float* b1   = (const float*)d_in[5];
    const float* W2l  = (const float*)d_in[6];
    const float* W2r  = (const float*)d_in[7];
    const float* att2 = (const float*)d_in[8];
    const float* b2   = (const float*)d_in[9];
    float* out = (float*)d_out;

    // workspace layout
    char* p = (char*)d_ws;
    size_t off = 0;
    auto carve = [&](size_t bytes) { void* r = p + off; off += align256(bytes); return r; };
    float*    xl1    = (float*)   carve((size_t)NN * 128 * 4);
    float*    xr1    = (float*)   carve((size_t)NN * 128 * 4);
    float*    hbuf   = (float*)   carve((size_t)NN * 128 * 4);   // accumulator, then h
    float*    score1 = (float*)   carve((size_t)E2 * HEADS * 4);
    unsigned* mkey1  = (unsigned*)carve((size_t)NN * HEADS * 4);
    float*    denom1 = (float*)   carve((size_t)NN * HEADS * 4);
    float*    xl2    = (float*)   carve((size_t)NN * CLASSES * 4);
    float*    xr2    = (float*)   carve((size_t)NN * CLASSES * 4);
    float*    score2 = (float*)   carve((size_t)E2 * 4);
    unsigned* mkey2  = (unsigned*)carve((size_t)NN * 4);
    float*    denom2 = (float*)   carve((size_t)NN * 4);
    // packed bf16 weight fragments (tiny, L2-resident)
    const int KS = F_IN / 32;                                    // 4 K-steps
    __bf16* Bp1l = (__bf16*)carve((size_t)8 * KS * 32 * 16 * 2);
    __bf16* Bp1r = (__bf16*)carve((size_t)8 * KS * 32 * 16 * 2);
    __bf16* Bp2l = (__bf16*)carve((size_t)1 * KS * 32 * 16 * 2);
    __bf16* Bp2r = (__bf16*)carve((size_t)1 * KS * 32 * 16 * 2);

    // ---- pack weights into WMMA fragment order ----
    {
        int tot1 = 8 * KS * 32 * 16, tot2 = 1 * KS * 32 * 16;
        pack_B<<<(tot1 + 255) / 256, 256, 0, stream>>>(W1l, Bp1l, HEADS * HID, 8, KS);
        pack_B<<<(tot1 + 255) / 256, 256, 0, stream>>>(W1r, Bp1r, HEADS * HID, 8, KS);
        pack_B<<<(tot2 + 255) / 256, 256, 0, stream>>>(W2l, Bp2l, CLASSES, 1, KS);
        pack_B<<<(tot2 + 255) / 256, 256, 0, stream>>>(W2r, Bp2r, CLASSES, 1, KS);
    }

    const int mtiles = NN / 16;                 // 6250
    const int gblocks = (mtiles + 7) / 8;       // one wave per M-tile, 8 waves/block
    // ---- layer 1 GEMMs ----
    gemm_bf16_wmma<8, F_IN><<<gblocks, 256, 0, stream>>>(x, (const v16bf*)Bp1l, xl1, NN, HEADS * HID, HEADS * HID);
    gemm_bf16_wmma<8, F_IN><<<gblocks, 256, 0, stream>>>(x, (const v16bf*)Bp1r, xr1, NN, HEADS * HID, HEADS * HID);

    hipMemsetAsync(mkey1,  0, (size_t)NN * HEADS * 4, stream);
    hipMemsetAsync(denom1, 0, (size_t)NN * HEADS * 4, stream);
    hipMemsetAsync(hbuf,   0, (size_t)NN * 128 * 4,   stream);

    // ---- layer 1 edge phase: one wave per edge ----
    {
        long long blocks = ((long long)E2 * 32 + 255) / 256;   // 8 waves / block
        edge_score_l1<<<(int)blocks, 256, 0, stream>>>(ei, xl1, xr1, att1, score1, mkey1);
        edge_agg_l1  <<<(int)blocks, 256, 0, stream>>>(ei, xl1, score1, mkey1, denom1, hbuf);
    }
    node_norm_l1<<<((size_t)NN * 128 + 255) / 256, 256, 0, stream>>>(hbuf, denom1, b1);

    // ---- layer 2 GEMMs: 1 column tile (10 cols, zero-padded in pack) ----
    gemm_bf16_wmma<1, HEADS * HID><<<gblocks, 256, 0, stream>>>(hbuf, (const v16bf*)Bp2l, xl2, NN, CLASSES, CLASSES);
    gemm_bf16_wmma<1, HEADS * HID><<<gblocks, 256, 0, stream>>>(hbuf, (const v16bf*)Bp2r, xr2, NN, CLASSES, CLASSES);

    hipMemsetAsync(mkey2,  0, (size_t)NN * 4, stream);
    hipMemsetAsync(denom2, 0, (size_t)NN * 4, stream);
    hipMemsetAsync(out,    0, (size_t)NN * CLASSES * 4, stream);

    // ---- layer 2 edge phase: thread per edge ----
    {
        int blocks = (E2 + 255) / 256;
        edge_score_l2<<<blocks, 256, 0, stream>>>(ei, xl2, xr2, att2, score2, mkey2);
        edge_agg_l2  <<<blocks, 256, 0, stream>>>(ei, xl2, score2, mkey2, denom2, out);
    }
    node_final<<<((size_t)NN * CLASSES + 255) / 256, 256, 0, stream>>>(out, denom2, b2);
}
